// InfiniAttention_27736898798183
// MI455X (gfx1250) — matrix operand order repaired and checked by reading, not compiled, not run on hardware
//
#include <hip/hip_runtime.h>

#define H_   16
#define NB_  4
#define D_   128
#define HID_ 2048
#define B_   4
#define EPS_ 1e-6f

typedef __attribute__((ext_vector_type(16))) __bf16 v16bf;
typedef __attribute__((ext_vector_type(8)))  float  v8f;

union Frag { v16bf v; uint4 q[2]; };

__device__ __forceinline__ unsigned short f2bf(float f) {
  unsigned int u = __float_as_uint(f);
  u += 0x7FFFu + ((u >> 16) & 1u);          // round-to-nearest-even
  return (unsigned short)(u >> 16);
}
__device__ __forceinline__ float bf2f(unsigned short s) {
  return __uint_as_float(((unsigned int)s) << 16);
}

// ---------------------------------------------------------------------------
// C[M,N] = A[M,K](bf16) @ W[N,K]^T(bf16), fp32 accumulate via WMMA bf16.
// epi: 0 = fp32 out, 1 = elu+1 -> bf16 out, 2 = bf16 out.
// Tile 128x128, K-step 32, 256 threads (8 waves), wave computes 32x64.
// Double-buffered LDS: one barrier per k-step; next tile staged in VGPRs
// while WMMAs consume the current LDS buffer.
// ---------------------------------------------------------------------------
__global__ __launch_bounds__(256) void gemm_bf16_nt(
    const unsigned short* __restrict__ A,
    const unsigned short* __restrict__ Wt,
    void* __restrict__ out, int M, int N, int K, int epi)
{
  __shared__ __align__(16) unsigned short As[2][128][32];
  __shared__ __align__(16) unsigned short Bs[2][128][32];
  const int tid  = threadIdx.x;
  const int lane = tid & 31;
  const int wave = tid >> 5;
  const int half = lane >> 4;
  const int l16  = lane & 15;
  const int m0 = blockIdx.x * 128;
  const int n0 = blockIdx.y * 128;
  const int wm = (wave & 3) * 32;
  const int wn = (wave >> 2) * 64;

  v8f acc[2][4];
  for (int i = 0; i < 2; i++)
    for (int j = 0; j < 4; j++)
      for (int r = 0; r < 8; r++) acc[i][j][r] = 0.f;

  const int lr = tid >> 1;        // 0..127 (tile row)
  const int lc = (tid & 1) * 16;  // 0 / 16 (k sub-offset)
  const unsigned short* gA = A  + (size_t)(m0 + lr) * K + lc;
  const unsigned short* gW = Wt + (size_t)(n0 + lr) * K + lc;

  { // preload k-tile 0 into buffer 0
    const uint4* sa = (const uint4*)gA;
    uint4* da = (uint4*)(&As[0][lr][lc]);
    da[0] = sa[0]; da[1] = sa[1];
    const uint4* sb = (const uint4*)gW;
    uint4* db = (uint4*)(&Bs[0][lr][lc]);
    db[0] = sb[0]; db[1] = sb[1];
  }
  __syncthreads();

  int buf = 0;
  for (int k0 = 0; k0 < K; k0 += 32, buf ^= 1) {
    const bool has_next = (k0 + 32) < K;
    uint4 ra0, ra1, rb0, rb1;
    if (has_next) { // stage next k-tile into registers (overlaps with WMMA)
      const uint4* sa = (const uint4*)(gA + k0 + 32);
      ra0 = sa[0]; ra1 = sa[1];
      const uint4* sb = (const uint4*)(gW + k0 + 32);
      rb0 = sb[0]; rb1 = sb[1];
      if (k0 + 64 < K) {
        __builtin_prefetch(gA + k0 + 64, 0, 1);
        __builtin_prefetch(gW + k0 + 64, 0, 1);
      }
    }

    Frag af[2], bfr[4];
    #pragma unroll
    for (int i = 0; i < 2; i++) {
      const uint4* p = (const uint4*)(&As[buf][wm + i * 16 + l16][0]);
      af[i].q[0] = p[half];          // k = 8*half .. +7
      af[i].q[1] = p[2 + half];      // k = 16+8*half .. +7
    }
    #pragma unroll
    for (int j = 0; j < 4; j++) {
      const uint4* p = (const uint4*)(&Bs[buf][wn + j * 16 + l16][0]);
      bfr[j].q[0] = p[half];         // k = 8*half .. +7
      bfr[j].q[1] = p[2 + half];     // k = 16+8*half .. +7
    }
    #pragma unroll
    for (int i = 0; i < 2; i++)
      #pragma unroll
      for (int j = 0; j < 4; j++)
        acc[i][j] = __builtin_amdgcn_wmma_f32_16x16x32_bf16(
            false, af[i].v, false, bfr[j].v, (short)0, acc[i][j], false, false);

    if (has_next) { // spill staged tile into the other buffer
      uint4* da = (uint4*)(&As[buf ^ 1][lr][lc]);
      da[0] = ra0; da[1] = ra1;
      uint4* db = (uint4*)(&Bs[buf ^ 1][lr][lc]);
      db[0] = rb0; db[1] = rb1;
    }
    __syncthreads();
  }

  for (int i = 0; i < 2; i++)
    for (int j = 0; j < 4; j++)
      for (int r = 0; r < 8; r++) {
        const int m = m0 + wm + i * 16 + r + 8 * half;
        const int n = n0 + wn + j * 16 + l16;
        const float c = acc[i][j][r];
        if (epi == 0) {
          ((float*)out)[(size_t)m * N + n] = c;
        } else if (epi == 1) {
          const float s = (c > 0.f) ? (c + 1.f) : __expf(c);   // elu(x)+1
          ((unsigned short*)out)[(size_t)m * N + n] = f2bf(s);
        } else {
          ((unsigned short*)out)[(size_t)m * N + n] = f2bf(c);
        }
      }
}

// ---------------------------------------------------------------------------
// combined[m, h*D+e] = sum_n weff[h,n] * (sigma_q[m] @ mem[n,h]) / clip(norm)
// One block = one head x 64-row tile; 4 waves, wave owns 32 e-columns.
// memT is [NB,H,E,D] (pre-transposed) so B-fragments read contiguous d-runs.
// ---------------------------------------------------------------------------
__global__ __launch_bounds__(128) void retrieve_banks(
    const unsigned short* __restrict__ SQ,
    const unsigned short* __restrict__ memT,
    const float* __restrict__ mem_norms,   // [NB,H,D]
    const float* __restrict__ weff,        // [H,NB] softmax*active
    unsigned short* __restrict__ Cb)
{
  __shared__ __align__(16) unsigned short Aq[64][128];
  __shared__ float nrm[NB_][64];
  const int h  = blockIdx.y;
  const int m0 = blockIdx.x * 64;
  const int tid = threadIdx.x, lane = tid & 31, wave = tid >> 5;
  const int half = lane >> 4, l16 = lane & 15;

  { // stage sigma_q tile
    const int m = tid & 63;
    const int c = (tid >> 6) * 64;
    const uint4* src = (const uint4*)(SQ + (size_t)(m0 + m) * HID_ + h * D_ + c);
    uint4* dst = (uint4*)(&Aq[m][c]);
    #pragma unroll
    for (int i = 0; i < 8; i++) dst[i] = src[i];
  }
  __syncthreads();
  for (int p = tid; p < NB_ * 64; p += 128) {  // per-row bank norms
    const int n = p >> 6, m = p & 63;
    const float* nv = mem_norms + ((size_t)n * H_ + h) * D_;
    float s = 0.f;
    for (int d = 0; d < D_; d++) s += bf2f(Aq[m][d]) * nv[d];
    nrm[n][m] = fmaxf(s, EPS_);
  }
  __syncthreads();

  const int wn = wave * 32;
  v8f comb[4][2];
  for (int i = 0; i < 4; i++)
    for (int j = 0; j < 2; j++)
      for (int r = 0; r < 8; r++) comb[i][j][r] = 0.f;

  for (int n = 0; n < NB_; n++) {
    const float w = weff[h * NB_ + n];
    const unsigned short* Bmat = memT + ((size_t)n * H_ + h) * (D_ * D_);
    v8f acc[4][2];
    for (int i = 0; i < 4; i++)
      for (int j = 0; j < 2; j++)
        for (int r = 0; r < 8; r++) acc[i][j][r] = 0.f;

    for (int k0 = 0; k0 < D_; k0 += 32) {
      Frag bfr[2];
      #pragma unroll
      for (int j = 0; j < 2; j++) {
        const uint4* p = (const uint4*)(Bmat + (size_t)(wn + j * 16 + l16) * D_ + k0);
        bfr[j].q[0] = p[half]; bfr[j].q[1] = p[2 + half];
      }
      #pragma unroll
      for (int i = 0; i < 4; i++) {
        const uint4* p = (const uint4*)(&Aq[i * 16 + l16][k0]);
        Frag af; af.q[0] = p[half]; af.q[1] = p[2 + half];
        #pragma unroll
        for (int j = 0; j < 2; j++)
          acc[i][j] = __builtin_amdgcn_wmma_f32_16x16x32_bf16(
              false, af.v, false, bfr[j].v, (short)0, acc[i][j], false, false);
      }
    }
    for (int i = 0; i < 4; i++)
      for (int j = 0; j < 2; j++)
        for (int r = 0; r < 8; r++) {
          const int m = i * 16 + r + 8 * half;
          comb[i][j][r] += w * acc[i][j][r] / nrm[n][m];
        }
  }
  for (int i = 0; i < 4; i++)
    for (int j = 0; j < 2; j++)
      for (int r = 0; r < 8; r++) {
        const int m = m0 + i * 16 + r + 8 * half;
        const int e = wn + j * 16 + l16;
        Cb[(size_t)m * HID_ + h * D_ + e] = f2bf(comb[i][j][r]);
      }
}

// ---------------------------------------------------------------------------
// delta_v = v - (sigma_k @ mem0)/clip(knorm); written bf16 in place over Vb.
// ---------------------------------------------------------------------------
__global__ __launch_bounds__(128) void delta_v_kernel(
    const unsigned short* __restrict__ SK,
    const unsigned short* __restrict__ memT,    // bank 0 used
    const float* __restrict__ mem_norms,        // bank 0 = first H*D
    unsigned short* __restrict__ Vb)
{
  __shared__ __align__(16) unsigned short Ak[64][128];
  __shared__ float knorm[64];
  const int h  = blockIdx.y;
  const int m0 = blockIdx.x * 64;
  const int tid = threadIdx.x, lane = tid & 31, wave = tid >> 5;
  const int half = lane >> 4, l16 = lane & 15;

  {
    const int m = tid & 63;
    const int c = (tid >> 6) * 64;
    const uint4* src = (const uint4*)(SK + (size_t)(m0 + m) * HID_ + h * D_ + c);
    uint4* dst = (uint4*)(&Ak[m][c]);
    #pragma unroll
    for (int i = 0; i < 8; i++) dst[i] = src[i];
  }
  __syncthreads();
  if (tid < 64) {
    const float* nv = mem_norms + (size_t)h * D_;
    float s = 0.f;
    for (int d = 0; d < D_; d++) s += bf2f(Ak[tid][d]) * nv[d];
    knorm[tid] = fmaxf(s, EPS_);
  }
  __syncthreads();

  const int wn = wave * 32;
  const unsigned short* Bmat = memT + (size_t)h * (D_ * D_);
  v8f acc[4][2];
  for (int i = 0; i < 4; i++)
    for (int j = 0; j < 2; j++)
      for (int r = 0; r < 8; r++) acc[i][j][r] = 0.f;

  for (int k0 = 0; k0 < D_; k0 += 32) {
    Frag bfr[2];
    #pragma unroll
    for (int j = 0; j < 2; j++) {
      const uint4* p = (const uint4*)(Bmat + (size_t)(wn + j * 16 + l16) * D_ + k0);
      bfr[j].q[0] = p[half]; bfr[j].q[1] = p[2 + half];
    }
    #pragma unroll
    for (int i = 0; i < 4; i++) {
      const uint4* p = (const uint4*)(&Ak[i * 16 + l16][k0]);
      Frag af; af.q[0] = p[half]; af.q[1] = p[2 + half];
      #pragma unroll
      for (int j = 0; j < 2; j++)
        acc[i][j] = __builtin_amdgcn_wmma_f32_16x16x32_bf16(
            false, af.v, false, bfr[j].v, (short)0, acc[i][j], false, false);
    }
  }
  for (int i = 0; i < 4; i++)
    for (int j = 0; j < 2; j++)
      for (int r = 0; r < 8; r++) {
        const int ml = i * 16 + r + 8 * half;
        const int e  = wn + j * 16 + l16;
        const size_t idx = (size_t)(m0 + ml) * HID_ + h * D_ + e;
        const float dv = bf2f(Vb[idx]) - acc[i][j][r] / knorm[ml];
        Vb[idx] = f2bf(dv);
      }
}

// ---------------------------------------------------------------------------
// mem_update[h,d,e] += sum_m sigma_k[m,h,d]*delta_v[m,h,e]  (partial, atomics)
// Block = (head, 1024-row chunk); 8 waves, wave owns 16 d-rows x 128 e-cols.
// ---------------------------------------------------------------------------
__global__ __launch_bounds__(256) void mem_update_kernel(
    const unsigned short* __restrict__ SK,
    const unsigned short* __restrict__ DV,
    float* __restrict__ macc, int Mchunk)
{
  __shared__ __align__(16) unsigned short Ad[128][32]; // [d][mm]
  __shared__ __align__(16) unsigned short Be[128][32]; // [e][mm]
  const int h  = blockIdx.y;
  const int m0 = blockIdx.x * Mchunk;
  const int tid = threadIdx.x, lane = tid & 31, wave = tid >> 5;
  const int half = lane >> 4, l16 = lane & 15;
  const int wd = wave * 16;

  v8f acc[8];
  for (int j = 0; j < 8; j++)
    for (int r = 0; r < 8; r++) acc[j][r] = 0.f;

  const int qi  = tid & 15;   // uint4 index along d/e (8 elems each)
  const int mmb = tid >> 4;   // 0..15

  for (int mb = 0; mb < Mchunk; mb += 32) {
    #pragma unroll
    for (int pass = 0; pass < 2; pass++) {        // transpose-stage into LDS
      const int mm = mmb + pass * 16;
      const size_t base = (size_t)(m0 + mb + mm) * HID_ + (size_t)h * D_ + qi * 8;
      const uint4 va = *(const uint4*)(SK + base);
      const uint4 vb = *(const uint4*)(DV + base);
      const unsigned short* pa = (const unsigned short*)&va;
      const unsigned short* pb = (const unsigned short*)&vb;
      #pragma unroll
      for (int x = 0; x < 8; x++) { Ad[qi * 8 + x][mm] = pa[x]; Be[qi * 8 + x][mm] = pb[x]; }
    }
    __syncthreads();
    Frag af;
    { const uint4* p = (const uint4*)(&Ad[wd + l16][0]);
      af.q[0] = p[half]; af.q[1] = p[2 + half]; }
    #pragma unroll
    for (int j = 0; j < 8; j++) {
      Frag bfr;
      const uint4* p = (const uint4*)(&Be[j * 16 + l16][0]);
      bfr.q[0] = p[half]; bfr.q[1] = p[2 + half];
      acc[j] = __builtin_amdgcn_wmma_f32_16x16x32_bf16(
          false, af.v, false, bfr.v, (short)0, acc[j], false, false);
    }
    __syncthreads();
  }
  for (int j = 0; j < 8; j++)
    for (int r = 0; r < 8; r++) {
      const int d = wd + r + 8 * half;
      const int e = j * 16 + l16;
      atomicAdd(&macc[((size_t)h * D_ + d) * D_ + e], acc[j][r]);
    }
}

// ------------------------- small helper kernels ----------------------------
__global__ void conv_bf16(const float* __restrict__ s, unsigned short* __restrict__ d, long long n) {
  const long long i = (long long)blockIdx.x * blockDim.x + threadIdx.x;
  if (i < n) d[i] = f2bf(s[i]);
}
__global__ void conv_memT(const float* __restrict__ s, unsigned short* __restrict__ d, int n) {
  const int i = blockIdx.x * blockDim.x + threadIdx.x;
  if (i < n) {
    const int nh = i >> 14, rem = i & 16383;
    const int dd = rem >> 7, e = rem & 127;
    d[((size_t)nh << 14) + (e << 7) + dd] = f2bf(s[i]);   // [nh][e][d]
  }
}
__global__ void zero_f32(float* p, int n) {
  const int i = blockIdx.x * blockDim.x + threadIdx.x;
  if (i < n) p[i] = 0.f;
}
__global__ void finalize_mem(const float* __restrict__ mem0, const float* __restrict__ macc,
                             float* __restrict__ out, int n, float scale) {
  const int i = blockIdx.x * blockDim.x + threadIdx.x;
  if (i < n) out[i] = mem0[i] + macc[i] * scale;
}
__global__ void prep_weights(const float* __restrict__ bw, const float* __restrict__ mn,
                             float* __restrict__ weff) {
  const int t = threadIdx.x;
  if (t >= H_ * NB_) return;
  const int h = t >> 2, n = t & 3;
  float mx = bw[h * NB_];
  for (int j = 1; j < NB_; j++) mx = fmaxf(mx, bw[h * NB_ + j]);
  float den = 0.f;
  for (int j = 0; j < NB_; j++) den += __expf(bw[h * NB_ + j] - mx);
  const float w = __expf(bw[h * NB_ + n] - mx) / den;
  float s = 0.f;
  for (int i = 0; i < H_ * D_; i++) s += mn[(size_t)n * H_ * D_ + i];
  weff[t] = (s >= EPS_) ? w : 0.f;
}
__global__ __launch_bounds__(256) void norm_reduce(
    const unsigned short* __restrict__ SK, const float* __restrict__ mem_norms,
    float* __restrict__ out, int M) {
  __shared__ float red[256];
  const int hd = blockIdx.x;
  float s = 0.f;
  for (int m = threadIdx.x; m < M; m += 256) s += bf2f(SK[(size_t)m * HID_ + hd]);
  red[threadIdx.x] = s;
  __syncthreads();
  for (int o = 128; o > 0; o >>= 1) {
    if (threadIdx.x < o) red[threadIdx.x] += red[threadIdx.x + o];
    __syncthreads();
  }
  if (threadIdx.x == 0) out[hd] = mem_norms[hd] + red[0] * (1.0f / B_);
}

// ---------------------------------------------------------------------------
extern "C" void kernel_launch(void* const* d_in, const int* in_sizes, int n_in,
                              void* d_out, int out_size, void* d_ws, size_t ws_size,
                              hipStream_t stream) {
  const float* hidden    = (const float*)d_in[0];
  const float* Wq        = (const float*)d_in[1];
  const float* Wk        = (const float*)d_in[2];
  const float* Wv        = (const float*)d_in[3];
  const float* Wo        = (const float*)d_in[4];
  const float* bank_w    = (const float*)d_in[5];
  const float* memories  = (const float*)d_in[6];
  const float* mem_norms = (const float*)d_in[7];

  const int M = in_sizes[0] / HID_;   // B*S = 16384

  char* ws = (char*)d_ws;
  size_t off = 0;
  auto take = [&](size_t bytes) -> char* {
    char* p = ws + off;
    off += (bytes + 255) & ~(size_t)255;
    return p;
  };
  unsigned short* Xb    = (unsigned short*)take((size_t)M * HID_ * 2);
  unsigned short* Wqb   = (unsigned short*)take((size_t)HID_ * HID_ * 2);
  unsigned short* Wkb   = (unsigned short*)take((size_t)HID_ * HID_ * 2);
  unsigned short* Wvb   = (unsigned short*)take((size_t)HID_ * HID_ * 2);
  unsigned short* Wob   = (unsigned short*)take((size_t)HID_ * HID_ * 2);
  unsigned short* SQ    = (unsigned short*)take((size_t)M * HID_ * 2);
  unsigned short* SK    = (unsigned short*)take((size_t)M * HID_ * 2);
  unsigned short* Vb    = (unsigned short*)take((size_t)M * HID_ * 2);
  unsigned short* Cb    = (unsigned short*)take((size_t)M * HID_ * 2);
  unsigned short* memTb = (unsigned short*)take((size_t)NB_ * H_ * D_ * D_ * 2);
  float*          weff  = (float*)take((size_t)H_ * NB_ * 4);
  float*          macc  = (float*)take((size_t)H_ * D_ * D_ * 4);

  float* out_attn = (float*)d_out;                       // [M, HID]
  float* out_mem  = out_attn + (size_t)M * HID_;         // [H, D, D]
  float* out_norm = out_mem + (size_t)H_ * D_ * D_;      // [H, D]

  // --- precision conversion / layout prep ---
  {
    const long long nx = (long long)M * HID_;
    conv_bf16<<<dim3((unsigned)((nx + 255) / 256)), 256, 0, stream>>>(hidden, Xb, nx);
    const long long nw = (long long)HID_ * HID_;
    conv_bf16<<<dim3((unsigned)((nw + 255) / 256)), 256, 0, stream>>>(Wq, Wqb, nw);
    conv_bf16<<<dim3((unsigned)((nw + 255) / 256)), 256, 0, stream>>>(Wk, Wkb, nw);
    conv_bf16<<<dim3((unsigned)((nw + 255) / 256)), 256, 0, stream>>>(Wv, Wvb, nw);
    conv_bf16<<<dim3((unsigned)((nw + 255) / 256)), 256, 0, stream>>>(Wo, Wob, nw);
    const int nm = NB_ * H_ * D_ * D_;
    conv_memT<<<dim3((nm + 255) / 256), 256, 0, stream>>>(memories, memTb, nm);
    prep_weights<<<1, 64, 0, stream>>>(bank_w, mem_norms, weff);
  }

  // --- projections (bf16 WMMA, fused elu+1 epilogues) ---
  dim3 gg(M / 128, HID_ / 128);
  gemm_bf16_nt<<<gg, 256, 0, stream>>>(Xb, Wqb, SQ, M, HID_, HID_, 1); // sigma_q
  gemm_bf16_nt<<<gg, 256, 0, stream>>>(Xb, Wkb, SK, M, HID_, HID_, 1); // sigma_k
  gemm_bf16_nt<<<gg, 256, 0, stream>>>(Xb, Wvb, Vb, M, HID_, HID_, 2); // v

  // --- bank retrieval + delta rule ---
  retrieve_banks<<<dim3(M / 64, H_), 128, 0, stream>>>(SQ, memTb, mem_norms, weff, Cb);
  delta_v_kernel<<<dim3(M / 64, H_), 128, 0, stream>>>(SK, memTb, mem_norms, Vb);

  // --- memory update reduction ---
  zero_f32<<<dim3((H_ * D_ * D_ + 255) / 256), 256, 0, stream>>>(macc, H_ * D_ * D_);
  mem_update_kernel<<<dim3(M / 1024, H_), 256, 0, stream>>>(SK, Vb, macc, 1024);
  finalize_mem<<<dim3((H_ * D_ * D_ + 255) / 256), 256, 0, stream>>>(
      memories, macc, out_mem, H_ * D_ * D_, 1.0f / (float)M);
  norm_reduce<<<dim3(H_ * D_), 256, 0, stream>>>(SK, mem_norms, out_norm, M);

  // --- output projection straight into d_out (fp32 epilogue) ---
  gemm_bf16_nt<<<gg, 256, 0, stream>>>(Cb, Wob, out_attn, M, HID_, HID_, 0);
}
